// vGINMolEncoder_2654289789377
// MI455X (gfx1250) — matrix operand order repaired and checked by reading, not compiled, not run on hardware
//
#include <hip/hip_runtime.h>

typedef __attribute__((ext_vector_type(4)))  _Float16 v4h;
typedef __attribute__((ext_vector_type(8)))  _Float16 v8h;
typedef __attribute__((ext_vector_type(16))) _Float16 v16h;
typedef __attribute__((ext_vector_type(8)))  float    v8f;

#define Nn 100000
#define Ee 400000
#define Bb 2048
#define Dd 300
#define DP 320      // 300 padded to multiple of 32
#define Hh 600
#define HP 640      // 600 padded to multiple of 32
#define Ll 3

#define LDS_STRIDE 40            // halves per A-tile row (padded: conflict-free b128 reads)
#define LDS_BUF    (32 * LDS_STRIDE)

// ---------------------------------------------------------------------------
// WMMA GEMM: out[M, ldo] = epi( A[M,Kp](fp32) x WT[Ncp,Kp](f16)^T )
// epi(v) = maybe_relu(v * cs[col] + csh[col])
// block = 256 threads = 8 waves; block tile 32(M) x 64(N); wave tile 16x16.
// A tile (32x32) staged fp32->f16 into LDS once per block, double-buffered.
// grid = (Ncp/64, M/32). Requires M%32==0, Kp%32==0, Ncp%64==0.
// ---------------------------------------------------------------------------
__global__ __launch_bounds__(256) void wmma_gemm_kernel(
    const float* __restrict__ A, int Kp,
    const _Float16* __restrict__ WT,
    const float* __restrict__ cs, const float* __restrict__ csh,
    float* __restrict__ out, int ldo, int relu)
{
    __shared__ _Float16 smem[2 * LDS_BUF];

    const int tid  = threadIdx.x;
    const int lane = tid & 31;
    const int wave = tid >> 5;
    const int wm   = wave >> 2;           // 0..1
    const int wn   = wave & 3;            // 0..3
    const int l16  = lane & 15;
    const int lhi  = lane >> 4;           // 0 or 1

    const int blockRow = blockIdx.y * 32;
    const int colbase  = blockIdx.x * 64 + wn * 16;

    // --- staging role: thread t converts 4 fp32 -> 4 f16 of the 32x32 A tile
    const int srow = tid >> 3;            // 0..31
    const int skg  = (tid & 7) * 4;       // 0,4,...,28
    const float*    aSrc  = A + (size_t)(blockRow + srow) * Kp + skg;
    _Float16*       sDst  = smem + srow * LDS_STRIDE + skg;

    // --- consumption pointers
    // A frag (from LDS): row = wm*16+l16; elems 0..15 -> k=lhi*16..lhi*16+15
    const _Float16* aFrag = smem + (wm * 16 + l16) * LDS_STRIDE + lhi * 16;
    // B frag (global): col = colbase+l16; 16 contiguous K at kb + lhi*16
    const _Float16* brow  = WT + (size_t)(colbase + l16) * Kp + lhi * 16;

    // stage first K tile into buffer 0
    {
        float4 v = *(const float4*)(aSrc);
        v4h h;
        h.x = (_Float16)v.x; h.y = (_Float16)v.y; h.z = (_Float16)v.z; h.w = (_Float16)v.w;
        *(v4h*)sDst = h;
    }

    v8f acc = {};
    int ib = 0;
    for (int kb = 0; kb < Kp; kb += 32, ib ^= 1) {
        __syncthreads();
        if (kb + 32 < Kp) {               // prefetch next K tile into other buffer
            float4 v = *(const float4*)(aSrc + kb + 32);
            v4h h;
            h.x = (_Float16)v.x; h.y = (_Float16)v.y; h.z = (_Float16)v.z; h.w = (_Float16)v.w;
            *(v4h*)(sDst + (ib ^ 1) * LDS_BUF) = h;
        }
        const _Float16* ap = aFrag + ib * LDS_BUF;
        v8h alo = *(const v8h*)(ap);          // k = lhi*16 .. +7
        v8h ahi = *(const v8h*)(ap + 8);      // k = lhi*16+8 .. +15
        v16h af = __builtin_shufflevector(alo, ahi,
                                          0, 1, 2, 3, 4, 5, 6, 7,
                                          8, 9, 10, 11, 12, 13, 14, 15);
        v16h bf = *(const v16h*)(brow + kb);
        acc = __builtin_amdgcn_wmma_f32_16x16x32_f16(
            false, af, false, bf, (short)0, acc, false, false);
    }

    const int col = colbase + l16;
    const float s  = cs[col];
    const float sh = csh[col];
    #pragma unroll
    for (int r = 0; r < 8; ++r) {
        const int row = blockRow + wm * 16 + r + lhi * 8;  // D: M = r + 8*(lane/16)
        float v = fmaf(acc[r], s, sh);
        if (relu) v = fmaxf(v, 0.f);
        out[(size_t)row * ldo + col] = v;
    }
}

// ---------------------------------------------------------------------------
// Weight pre-pack: W[Kreal,Ncreal] fp32 row-major -> WT[Ncp,Kp] f16 (transposed,
// zero-padded).
// ---------------------------------------------------------------------------
__global__ void convert_wt_kernel(const float* __restrict__ W, int Kreal, int Ncreal,
                                  _Float16* __restrict__ WT, int Kp, int Ncp)
{
    int idx = blockIdx.x * blockDim.x + threadIdx.x;
    int total = Kp * Ncp;
    if (idx >= total) return;
    int n = idx / Kp;
    int k = idx - n * Kp;
    float v = (n < Ncreal && k < Kreal) ? W[(size_t)k * Ncreal + n] : 0.f;
    WT[idx] = (_Float16)v;
}

// Fused BN+bias params: cs = g/sqrt(v+eps), csh = (bias-m)*cs + b ; zero pad tail.
__global__ void pack_params_kernel(const float* __restrict__ g, const float* __restrict__ b,
                                   const float* __restrict__ m, const float* __restrict__ v,
                                   const float* __restrict__ bias, int nreal, int np,
                                   float* __restrict__ cs, float* __restrict__ csh)
{
    int i = blockIdx.x * blockDim.x + threadIdx.x;
    if (i >= np) return;
    if (i < nreal) {
        float s = g[i] * rsqrtf(v[i] + 1e-5f);
        cs[i]  = s;
        csh[i] = (bias[i] - m[i]) * s + b[i];
    } else {
        cs[i] = 0.f; csh[i] = 0.f;
    }
}

// h[n,c] = sum_j atom_emb[j, x[n,j], c]   (c<300, else 0); X stride DP
__global__ void atom_encode_kernel(const int* __restrict__ x,
                                   const float* __restrict__ atom_emb,
                                   float* __restrict__ X)
{
    int n = blockIdx.x;
    int c = threadIdx.x;            // 0..319
    float acc = 0.f;
    if (c < Dd) {
        #pragma unroll
        for (int j = 0; j < 9; ++j) {
            int a = x[n * 9 + j];
            acc += atom_emb[((size_t)j * 64 + a) * Dd + c];
        }
    }
    X[(size_t)n * DP + c] = acc;
}

__global__ void init_vfeat_kernel(const float* __restrict__ vemb, float* __restrict__ vfeat)
{
    int b = blockIdx.x;
    int c = threadIdx.x;
    vfeat[(size_t)b * DP + c] = (c < Dd) ? vemb[c] : 0.f;
}

// msg = relu(X[src] + bond_embed); atomicAdd into aggr[dst]
__global__ void edge_msg_kernel(const int* __restrict__ ei, const int* __restrict__ ea,
                                const float* __restrict__ bond,   // bond_emb + l*3*8*Dd
                                const float* __restrict__ X, float* __restrict__ aggr)
{
    int e = blockIdx.x;
    int c = threadIdx.x;
    if (c >= Dd) return;
    int src = ei[e];
    int dst = ei[Ee + e];
    int a0 = ea[e * 3 + 0];
    int a1 = ea[e * 3 + 1];
    int a2 = ea[e * 3 + 2];
    float ev = bond[(size_t)(0 * 8 + a0) * Dd + c]
             + bond[(size_t)(1 * 8 + a1) * Dd + c]
             + bond[(size_t)(2 * 8 + a2) * Dd + c];
    float msg = X[(size_t)src * DP + c] + ev;
    msg = fmaxf(msg, 0.f);
    atomicAdd(&aggr[(size_t)dst * DP + c], msg);
}

// Z = (1+eps[l]) * X + aggr   (elementwise over N*DP)
__global__ void z_combine_kernel(const float* __restrict__ X, const float* __restrict__ aggr,
                                 const float* __restrict__ eps_arr, int l,
                                 float* __restrict__ Z, int total)
{
    int i = blockIdx.x * blockDim.x + threadIdx.x;
    if (i >= total) return;
    float eps = eps_arr[l];
    Z[i] = (1.f + eps) * X[i] + aggr[i];
}

// X[n,:] += vfeat[batch[n],:]
__global__ void add_vfeat_kernel(float* __restrict__ X, const float* __restrict__ vfeat,
                                 const int* __restrict__ batch)
{
    int n = blockIdx.x;
    int c = threadIdx.x;
    X[(size_t)n * DP + c] += vfeat[(size_t)batch[n] * DP + c];
}

// segment-sum: out[batch[n], c] += X[n, c], c < 300
__global__ void pool_nodes_kernel(const float* __restrict__ X, const int* __restrict__ batch,
                                  float* __restrict__ out, int ldo)
{
    int n = blockIdx.x;
    int c = threadIdx.x;
    if (c >= Dd) return;
    atomicAdd(&out[(size_t)batch[n] * ldo + c], X[(size_t)n * DP + c]);
}

// vsum = pooledv + vfeat (elementwise B*DP)
__global__ void vsum_add_kernel(const float* __restrict__ a, const float* __restrict__ b,
                                float* __restrict__ o, int total)
{
    int i = blockIdx.x * blockDim.x + threadIdx.x;
    if (i >= total) return;
    o[i] = a[i] + b[i];
}

// ---------------------------------------------------------------------------
extern "C" void kernel_launch(void* const* d_in, const int* in_sizes, int n_in,
                              void* d_out, int out_size, void* d_ws, size_t ws_size,
                              hipStream_t stream)
{
    (void)in_sizes; (void)n_in; (void)ws_size;

    const int*   x          = (const int*)  d_in[0];
    const int*   edge_index = (const int*)  d_in[1];
    const int*   edge_attr  = (const int*)  d_in[2];
    const int*   batch      = (const int*)  d_in[3];
    const float* atom_emb   = (const float*)d_in[5];
    const float* vemb       = (const float*)d_in[6];
    const float* bond_emb   = (const float*)d_in[7];
    const float* conv_w1    = (const float*)d_in[8];
    const float* conv_b1    = (const float*)d_in[9];
    const float* cbn_g      = (const float*)d_in[10];
    const float* cbn_b      = (const float*)d_in[11];
    const float* cbn_m      = (const float*)d_in[12];
    const float* cbn_v      = (const float*)d_in[13];
    const float* conv_w2    = (const float*)d_in[14];
    const float* conv_b2    = (const float*)d_in[15];
    const float* conv_eps   = (const float*)d_in[16];
    const float* bn_g       = (const float*)d_in[17];
    const float* bn_b       = (const float*)d_in[18];
    const float* bn_m       = (const float*)d_in[19];
    const float* bn_v       = (const float*)d_in[20];
    const float* vw1        = (const float*)d_in[21];
    const float* vb1        = (const float*)d_in[22];
    const float* vbn1_g     = (const float*)d_in[23];
    const float* vbn1_b     = (const float*)d_in[24];
    const float* vbn1_m     = (const float*)d_in[25];
    const float* vbn1_v     = (const float*)d_in[26];
    const float* vw2        = (const float*)d_in[27];
    const float* vb2        = (const float*)d_in[28];
    const float* vbn2_g     = (const float*)d_in[29];
    const float* vbn2_b     = (const float*)d_in[30];
    const float* vbn2_m     = (const float*)d_in[31];
    const float* vbn2_v     = (const float*)d_in[32];

    char*  base = (char*)d_ws;
    size_t off  = 0;
    auto carve = [&](size_t bytes) -> void* {
        void* p = base + off;
        off += (bytes + 255) & ~(size_t)255;
        return p;
    };

    float* X       = (float*)carve((size_t)Nn * DP * 4);   // node features (K-padded)
    float* Z       = (float*)carve((size_t)Nn * DP * 4);   // (1+eps)X + aggr
    float* big     = (float*)carve((size_t)Nn * HP * 4);   // aggr (front) / mid (whole), aliased
    float* aggr    = big;
    float* mid     = big;
    float* vfeat   = (float*)carve((size_t)Bb * DP * 4);
    float* pooledv = (float*)carve((size_t)Bb * DP * 4);
    float* vsum    = (float*)carve((size_t)Bb * DP * 4);
    float* vmid    = (float*)carve((size_t)Bb * HP * 4);
    _Float16* W1T  = (_Float16*)carve((size_t)Ll * HP * DP * 2);
    _Float16* W2T  = (_Float16*)carve((size_t)Ll * DP * HP * 2);
    _Float16* vW1T = (_Float16*)carve((size_t)HP * DP * 2);
    _Float16* vW2T = (_Float16*)carve((size_t)DP * HP * 2);
    float* cs1  = (float*)carve((size_t)Ll * HP * 4);
    float* csh1 = (float*)carve((size_t)Ll * HP * 4);
    float* cs2  = (float*)carve((size_t)Ll * DP * 4);
    float* csh2 = (float*)carve((size_t)Ll * DP * 4);
    float* vcs1  = (float*)carve((size_t)HP * 4);
    float* vcsh1 = (float*)carve((size_t)HP * 4);
    float* vcs2  = (float*)carve((size_t)DP * 4);
    float* vcsh2 = (float*)carve((size_t)DP * 4);

    // ---- pre-pack weights (f16, transposed, padded) + fused BN params -----
    const int nwt = HP * DP;                       // same element count both ways
    for (int l = 0; l < Ll; ++l) {
        convert_wt_kernel<<<(nwt + 255) / 256, 256, 0, stream>>>(
            conv_w1 + (size_t)l * Dd * Hh, Dd, Hh, W1T + (size_t)l * HP * DP, DP, HP);
        convert_wt_kernel<<<(nwt + 255) / 256, 256, 0, stream>>>(
            conv_w2 + (size_t)l * Hh * Dd, Hh, Dd, W2T + (size_t)l * DP * HP, HP, DP);
        pack_params_kernel<<<(HP + 255) / 256, 256, 0, stream>>>(
            cbn_g + l * Hh, cbn_b + l * Hh, cbn_m + l * Hh, cbn_v + l * Hh,
            conv_b1 + l * Hh, Hh, HP, cs1 + l * HP, csh1 + l * HP);
        pack_params_kernel<<<(DP + 255) / 256, 256, 0, stream>>>(
            bn_g + l * Dd, bn_b + l * Dd, bn_m + l * Dd, bn_v + l * Dd,
            conv_b2 + l * Dd, Dd, DP, cs2 + l * DP, csh2 + l * DP);
    }
    convert_wt_kernel<<<(nwt + 255) / 256, 256, 0, stream>>>(vw1, Dd, Hh, vW1T, DP, HP);
    convert_wt_kernel<<<(nwt + 255) / 256, 256, 0, stream>>>(vw2, Hh, Dd, vW2T, HP, DP);
    pack_params_kernel<<<(HP + 255) / 256, 256, 0, stream>>>(
        vbn1_g, vbn1_b, vbn1_m, vbn1_v, vb1, Hh, HP, vcs1, vcsh1);
    pack_params_kernel<<<(DP + 255) / 256, 256, 0, stream>>>(
        vbn2_g, vbn2_b, vbn2_m, vbn2_v, vb2, Dd, DP, vcs2, vcsh2);

    // ---- atom encoder + virtual-node init ---------------------------------
    atom_encode_kernel<<<Nn, DP, 0, stream>>>(x, atom_emb, X);
    init_vfeat_kernel<<<Bb, DP, 0, stream>>>(vemb, vfeat);

    const int zTotal  = Nn * DP;
    const dim3 g1(HP / 64, Nn / 32);       // GEMM1 over nodes
    const dim3 g2(DP / 64, Nn / 32);       // GEMM2 over nodes
    const dim3 vg1(HP / 64, Bb / 32);      // vnode GEMM1
    const dim3 vg2(DP / 64, Bb / 32);      // vnode GEMM2

    for (int l = 0; l < Ll; ++l) {
        if (l > 0)
            add_vfeat_kernel<<<Nn, DP, 0, stream>>>(X, vfeat, batch);

        hipMemsetAsync(aggr, 0, (size_t)Nn * DP * 4, stream);
        edge_msg_kernel<<<Ee, DP, 0, stream>>>(
            edge_index, edge_attr, bond_emb + (size_t)l * 3 * 8 * Dd, X, aggr);
        z_combine_kernel<<<(zTotal + 255) / 256, 256, 0, stream>>>(
            X, aggr, conv_eps, l, Z, zTotal);

        // mid = relu(cbn(Z @ W1 + b1))   [N, HP]   (mid aliases aggr; aggr is dead)
        wmma_gemm_kernel<<<g1, 256, 0, stream>>>(
            Z, DP, W1T + (size_t)l * HP * DP, cs1 + l * HP, csh1 + l * HP, mid, HP, 1);
        // X = maybe_relu(bn(mid @ W2 + b2))   [N, DP]
        const int relu2 = (l < 2) ? 1 : 0;
        wmma_gemm_kernel<<<g2, 256, 0, stream>>>(
            mid, HP, W2T + (size_t)l * DP * HP, cs2 + l * DP, csh2 + l * DP, X, DP, relu2);

        if (l == 1) {   // virtual-node MLP (runs once: l=1 of the conv loop)
            hipMemsetAsync(pooledv, 0, (size_t)Bb * DP * 4, stream);
            pool_nodes_kernel<<<Nn, DP, 0, stream>>>(X, batch, pooledv, DP);
            vsum_add_kernel<<<(Bb * DP + 255) / 256, 256, 0, stream>>>(
                pooledv, vfeat, vsum, Bb * DP);
            wmma_gemm_kernel<<<vg1, 256, 0, stream>>>(
                vsum, DP, vW1T, vcs1, vcsh1, vmid, HP, 1);
            wmma_gemm_kernel<<<vg2, 256, 0, stream>>>(
                vmid, HP, vW2T, vcs2, vcsh2, vfeat, DP, 1);
        }
    }

    // ---- readout: segment_sum(post) -> d_out [B, 300] ----------------------
    hipMemsetAsync(d_out, 0, (size_t)out_size * 4, stream);
    pool_nodes_kernel<<<Nn, DP, 0, stream>>>(X, batch, (float*)d_out, Dd);
}
